// FFILinearNaive_84121229459539
// MI455X (gfx1250) — matrix operand order repaired and checked by reading, not compiled, not run on hardware
//
#include <hip/hip_runtime.h>
#include <stdint.h>

// Problem constants (match reference).
#define B_DIM   2048
#define IN_DIM  4096
#define OUT_DIM 16384
#define FAN_IN  208

// GEMM tiling.
#define BM 128
#define BN 128
#define BK 64                   // two 16x16x32 WMMA K-steps per tile
#define NKT (IN_DIM / BK)       // 64 K-tiles
#define LDS_STRIDE 72           // 64 data bf16 + 8 pad (TDM pad: 4 dwords per 32 dwords)
#define BUF_ELEMS (BM * LDS_STRIDE)          // one A or B buffer (uint16 elems)
#define BUF_BYTES (BUF_ELEMS * 2)            // 18432 B
#define SMEM_ELEMS (6 * BUF_ELEMS)           // A[3] + B[3] (110592 B)

typedef __attribute__((ext_vector_type(16))) __bf16 v16bf;
typedef __attribute__((ext_vector_type(8)))  float  v8f;
typedef unsigned int v4u __attribute__((ext_vector_type(4)));
typedef int          v4i __attribute__((ext_vector_type(4)));
typedef int          v8i __attribute__((ext_vector_type(8)));

struct Frag32 { uint4 lo, hi; };

static __device__ __forceinline__ v16bf make_frag(uint4 lo, uint4 hi) {
    Frag32 f{lo, hi};
    return __builtin_bit_cast(v16bf, f);
}

// fp32 -> bf16, round-to-nearest-even.
static __device__ __forceinline__ uint32_t f2bf(float x) {
    uint32_t u = __float_as_uint(x);
    u += 0x7FFFu + ((u >> 16) & 1u);
    return u >> 16;  // low 16 bits valid
}

// ---------------------------------------------------------------------------
// TDM: async DMA of a 2D tile [128 rows x 64 bf16], row stride IN_DIM elems,
// global -> LDS, with LDS padding 4 dwords after every 32 dwords (row stride
// 72 bf16 in LDS). Tracked by TENSORcnt. D# packing per CDNA5 ISA §8.
// ---------------------------------------------------------------------------
static __device__ __forceinline__ void tdm_load_tile(uint32_t lds_byte_off,
                                                     uint64_t gaddr) {
    v4u g0;
    g0.x = 1u;                                   // count=1 (valid), user mode
    g0.y = lds_byte_off;                         // lds_addr (bytes)
    g0.z = (uint32_t)gaddr;                      // global_addr[31:0]
    g0.w = (uint32_t)((gaddr >> 32) & 0x01FFFFFFu) | (2u << 30);  // [56:32] | type=2

    v8i g1;
    g1[0] = (int)((1u << 16)      // data_size = 1 -> 2 bytes
                | (1u << 20)      // pad_enable
                | (4u << 22)      // pad_interval: code 4 -> every 32 dwords (128 B row)
                | (3u << 25));    // pad_amount: code 3 -> 4 dwords (16 B = 8 bf16)
    g1[1] = (int)(((uint32_t)IN_DIM & 0xFFFFu) << 16);   // tensor_dim0[15:0]
    g1[2] = (int)(((uint32_t)IN_DIM >> 16)               // tensor_dim0[31:16]
                | (((uint32_t)OUT_DIM & 0xFFFFu) << 16));// tensor_dim1[15:0]
    g1[3] = (int)(((uint32_t)OUT_DIM >> 16)              // tensor_dim1[31:16]
                | (64u << 16));                          // tile_dim0 = 64 elems
    g1[4] = (int)(128u);                                 // tile_dim1 = 128, tile_dim2 = 0
    g1[5] = (int)(IN_DIM);                               // tensor_dim0_stride[31:0]
    g1[6] = 0;                                           // stride0[47:32] | stride1[15:0]
    g1[7] = 0;                                           // stride1[47:16]

    const v4i z4 = {0, 0, 0, 0};
    const v8i z8 = {0, 0, 0, 0, 0, 0, 0, 0};
    __builtin_amdgcn_tensor_load_to_lds(g0, g1, z4, z4, z8, 0);
}

// ---------------------------------------------------------------------------
// Kernel 1: x fp32 [2048,4096] -> bf16 (8 elems per thread).
// ---------------------------------------------------------------------------
__global__ __launch_bounds__(256) void cvt_x_kernel(const float* __restrict__ x,
                                                    uint16_t* __restrict__ xb) {
    const size_t i = ((size_t)blockIdx.x * 256 + threadIdx.x) * 8;
    const float4 a = *(const float4*)(x + i);
    const float4 b = *(const float4*)(x + i + 4);
    uint4 p;
    p.x = f2bf(a.x) | (f2bf(a.y) << 16);
    p.y = f2bf(a.z) | (f2bf(a.w) << 16);
    p.z = f2bf(b.x) | (f2bf(b.y) << 16);
    p.w = f2bf(b.z) | (f2bf(b.w) << 16);
    *(uint4*)(xb + i) = p;
}

// ---------------------------------------------------------------------------
// Kernel 2: densify one output row per block (deterministic scatter-add).
// Thread t owns dense columns [16t, 16t+16); scans entries in order.
// ---------------------------------------------------------------------------
__global__ __launch_bounds__(256) void densify_kernel(const int*   __restrict__ mask,
                                                      const float* __restrict__ cw,
                                                      uint16_t*    __restrict__ Wb) {
    const int o   = blockIdx.x;
    const int tid = threadIdx.x;

    __shared__ float row[IN_DIM];
    __shared__ int   sidx[FAN_IN];
    __shared__ float sw[FAN_IN];

    for (int i = tid; i < IN_DIM; i += 256) row[i] = 0.0f;
    if (tid < FAN_IN) {
        sidx[tid] = mask[(size_t)o * FAN_IN + tid];
        sw[tid]   = cw[(size_t)o * FAN_IN + tid];
    }
    __syncthreads();

    for (int e = 0; e < FAN_IN; ++e) {
        const int c = sidx[e];                 // LDS broadcast read
        if ((c >> 4) == tid) row[c] += sw[e];  // each column owned by one thread
    }
    // Each thread converts only columns it wrote -> no barrier needed.
    const float* r = row + tid * 16;
    uint4 p0, p1;
    p0.x = f2bf(r[0])  | (f2bf(r[1])  << 16);
    p0.y = f2bf(r[2])  | (f2bf(r[3])  << 16);
    p0.z = f2bf(r[4])  | (f2bf(r[5])  << 16);
    p0.w = f2bf(r[6])  | (f2bf(r[7])  << 16);
    p1.x = f2bf(r[8])  | (f2bf(r[9])  << 16);
    p1.y = f2bf(r[10]) | (f2bf(r[11]) << 16);
    p1.z = f2bf(r[12]) | (f2bf(r[13]) << 16);
    p1.w = f2bf(r[14]) | (f2bf(r[15]) << 16);
    uint4* dst = (uint4*)(Wb + (size_t)o * IN_DIM + tid * 16);
    dst[0] = p0;
    dst[1] = p1;
}

// ---------------------------------------------------------------------------
// Kernel 3: C[2048,16384] = x_bf16 @ W_bf16^T + bias, fp32 accumulate.
// 256 threads = 8 waves (2 M x 4 N); wave tile 64x32 = 4x2 WMMA 16x16 tiles.
// TDM triple-buffered pipeline: wave 0 issues TENSOR_LOAD_TO_LDS for tile
// kt+1 while all waves compute tile kt; one barrier per K-tile.
// ---------------------------------------------------------------------------
__global__ __launch_bounds__(256) void gemm_kernel(const uint16_t* __restrict__ xb,
                                                   const uint16_t* __restrict__ Wb,
                                                   const float*    __restrict__ bias,
                                                   float*          __restrict__ out) {
    extern __shared__ __align__(16) uint16_t smem[];
    // Layout: A buffers b=0..2 at [b*BUF), B buffers at [(3+b)*BUF).

    const int tid  = threadIdx.x;
    const int bm   = blockIdx.y * BM;
    const int bn   = blockIdx.x * BN;
    const int lane = tid & 31;
    const int wave = tid >> 5;
    const int wm   = (wave >> 2) * 64;  // 0 or 64
    const int wn   = (wave & 3) * 32;   // 0, 32, 64, 96
    const int lmod = lane & 15;
    const int lhi  = lane >> 4;
    const bool leader = (wave == 0);    // wave-uniform

    // LDS byte offset of dynamic shared (after static LDS).
    const uint32_t lds0 = __builtin_amdgcn_groupstaticsize();
    const uint64_t gA0 = (uint64_t)(uintptr_t)(xb + (size_t)bm * IN_DIM);
    const uint64_t gB0 = (uint64_t)(uintptr_t)(Wb + (size_t)bn * IN_DIM);

    v8f acc[4][2];
#pragma unroll
    for (int mf = 0; mf < 4; ++mf)
#pragma unroll
        for (int nf = 0; nf < 2; ++nf)
#pragma unroll
            for (int v = 0; v < 8; ++v) acc[mf][nf][v] = 0.0f;

    auto issue = [&](int kt, int b) {   // DMA tile kt into buffer b (2 tensor ops)
        tdm_load_tile(lds0 + (uint32_t)b * BUF_BYTES, gA0 + (size_t)kt * BK * 2);
        tdm_load_tile(lds0 + (uint32_t)(3 + b) * BUF_BYTES, gB0 + (size_t)kt * BK * 2);
    };
    auto compute = [&](int b) {
        const uint16_t* const Asm = smem + b * BUF_ELEMS;
        const uint16_t* const Bsm = smem + (3 + b) * BUF_ELEMS;
#pragma unroll
        for (int ks = 0; ks < 2; ++ks) {
            v16bf bfrag[2];
#pragma unroll
            for (int nf = 0; nf < 2; ++nf) {
                const uint16_t* p =
                    Bsm + (wn + nf * 16 + lmod) * LDS_STRIDE + ks * 32 + lhi * 8;
                bfrag[nf] = make_frag(*(const uint4*)p, *(const uint4*)(p + 16));
            }
#pragma unroll
            for (int mf = 0; mf < 4; ++mf) {
                const uint16_t* p =
                    Asm + (wm + mf * 16 + lmod) * LDS_STRIDE + ks * 32 + lhi * 8;
                const v16bf afrag = make_frag(*(const uint4*)p, *(const uint4*)(p + 16));
#pragma unroll
                for (int nf = 0; nf < 2; ++nf) {
                    acc[mf][nf] = __builtin_amdgcn_wmma_f32_16x16x32_bf16(
                        false, afrag, false, bfrag[nf], (short)0, acc[mf][nf], false, false);
                }
            }
        }
    };

    // Prologue: tile 0 in flight.
    if (leader) issue(0, 0);

    // Steady state: issue kt+1 (buffer safe: all waves passed barrier kt-1 =>
    // finished reading buffer (kt+1)%3 at iteration kt-2), wait for tile kt
    // (in-order TENSORcnt: <=2 leaves tile kt+1's 2 ops outstanding), barrier,
    // compute. One barrier per K-tile; DMA fully overlapped with WMMA.
    for (int kt = 0; kt < NKT - 1; ++kt) {
        if (leader) {
            issue(kt + 1, (kt + 1) % 3);
            __builtin_amdgcn_s_wait_tensorcnt(2);
        }
        __syncthreads();
        compute(kt % 3);
    }
    if (leader) __builtin_amdgcn_s_wait_tensorcnt(0);
    __syncthreads();
    compute((NKT - 1) % 3);

    // Epilogue: C layout lane l, VGPR v -> (M = v + 8*(l>>4), N = l&15).
    // Non-temporal: keep the 128 MB streamed output from evicting W_bf16 in L2.
#pragma unroll
    for (int nf = 0; nf < 2; ++nf) {
        const int n = bn + wn + nf * 16 + lmod;
        const float bv = bias[n];
#pragma unroll
        for (int mf = 0; mf < 4; ++mf) {
#pragma unroll
            for (int v = 0; v < 8; ++v) {
                const int m = bm + wm + mf * 16 + v + lhi * 8;
                __builtin_nontemporal_store(acc[mf][nf][v] + bv,
                                            &out[(size_t)m * OUT_DIM + n]);
            }
        }
    }
}

// ---------------------------------------------------------------------------
// Launch. Workspace layout (requires ws_size >= 151 MB, fully overwritten):
//   [0, 16 MiB)        x_bf16  [2048][4096]
//   [16 MiB, +128 MiB) W_bf16  [16384][4096]
// ---------------------------------------------------------------------------
extern "C" void kernel_launch(void* const* d_in, const int* in_sizes, int n_in,
                              void* d_out, int out_size, void* d_ws, size_t ws_size,
                              hipStream_t stream) {
    const float* x    = (const float*)d_in[0];
    const int*   mask = (const int*)d_in[1];
    const float* cw   = (const float*)d_in[2];
    const float* bias = (const float*)d_in[3];
    float*       out  = (float*)d_out;

    uint16_t* xb = (uint16_t*)d_ws;
    uint16_t* Wb = (uint16_t*)((char*)d_ws + (size_t)B_DIM * IN_DIM * sizeof(uint16_t));

    cvt_x_kernel<<<(B_DIM * IN_DIM) / (256 * 8), 256, 0, stream>>>(x, xb);
    densify_kernel<<<OUT_DIM, 256, 0, stream>>>(mask, cw, Wb);

    dim3 grid(OUT_DIM / BN, B_DIM / BM);
    gemm_kernel<<<grid, 256, SMEM_ELEMS * sizeof(uint16_t), stream>>>(xb, Wb, bias, out);
}
